// Net_PointTransformer_16544214024416
// MI455X (gfx1250) — matrix operand order repaired and checked by reading, not compiled, not run on hardware
//
#include <hip/hip_runtime.h>
#include <hip/hip_bf16.h>
#include <math.h>

// ---------------------------------------------------------------------------
// Types for WMMA
// ---------------------------------------------------------------------------
typedef __attribute__((ext_vector_type(16))) __bf16        v16bf;
typedef __attribute__((ext_vector_type(8)))  unsigned int  v8ui;
typedef __attribute__((ext_vector_type(8)))  float         v8f;

__device__ __forceinline__ unsigned int f2bf_pack(float lo, float hi) {
    union { float f; unsigned int u; } a, b; a.f = lo; b.f = hi;
    unsigned int ua = a.u, ub = b.u;
    ua += 0x7FFFu + ((ua >> 16) & 1u);     // round-to-nearest-even
    ub += 0x7FFFu + ((ub >> 16) & 1u);
    return (ua >> 16) | (ub & 0xFFFF0000u);
}

// ---------------------------------------------------------------------------
// Generic GEMM:  C[M,N] = A[M,K] @ B[K,N]  (+bias) (+relu)
// Block = 128 threads (4 waves). Block tile = 64 rows x 64 cols.
// Wave w owns columns [16w,16w+16); keeps 4 accumulators (rows g*16..g*16+15).
// LDS tiles are stored in WMMA *fragment order*: one 32B row per lane, so the
// compute phase issues wide ds_load_b128 pairs and 4 v_wmma per K-step with a
// single shared B fragment.
// flags: bit0 = relu, bit1 = bias
// ---------------------------------------------------------------------------
__global__ __launch_bounds__(128)
void gemm_wmma_kernel(const float* __restrict__ A, const float* __restrict__ B,
                      const float* __restrict__ bias, float* __restrict__ C,
                      int M, int K, int N, int flags)
{
    // [group][lane][dword] : 16 bf16 per lane = 32 bytes
    __shared__ __align__(32) unsigned int sAf[4][32][8];
    __shared__ __align__(32) unsigned int sBf[4][32][8];

    const int tid  = threadIdx.x;
    const int wave = tid >> 5;
    const int lane = tid & 31;
    const int row0 = blockIdx.y * 64;
    const int col0 = blockIdx.x * 64;

    v8f acc0 = {0.f,0.f,0.f,0.f,0.f,0.f,0.f,0.f};
    v8f acc1 = acc0, acc2 = acc0, acc3 = acc0;

    for (int k0 = 0; k0 < K; k0 += 32) {
        // ---- stage A tile (64x32) into fragment layout -------------------
        // element (r, kk):  g=r>>4, rr=r&15
        //   lane = rr + ((kk>>3)&1 ? 16 : 0)
        //   i    = (kk&7) + (kk>=16 ? 8 : 0)     (pack K pairs into dwords)
        for (int e = tid; e < 64 * 16; e += 128) {     // 16 K-pairs per row
            int r  = e >> 4;
            int kk = (e & 15) << 1;
            int gr = row0 + r;
            int gk = k0 + kk;
            float lo = (gr < M && gk     < K) ? A[(size_t)gr * K + gk]     : 0.f;
            float hi = (gr < M && gk + 1 < K) ? A[(size_t)gr * K + gk + 1] : 0.f;
            int ln = (r & 15) + (((kk >> 3) & 1) ? 16 : 0);
            int w  = ((kk & 7) >> 1) + ((kk >= 16) ? 4 : 0);
            sAf[r >> 4][ln][w] = f2bf_pack(lo, hi);
        }
        // ---- stage B tile (32x64) into fragment layout -------------------
        // element (kk, col): lane = (col&15) + ((kk>>3)&1 ? 16 : 0), i = (kk&7) + (kk>=16 ? 8 : 0)
        for (int e = tid; e < 16 * 64; e += 128) {     // 16 K-pairs x 64 cols
            int kp  = e >> 6;
            int col = e & 63;
            int kk  = kp << 1;
            int gk  = k0 + kk;
            int gc  = col0 + col;
            float lo = (gk     < K && gc < N) ? B[(size_t)gk * N + gc]       : 0.f;
            float hi = (gk + 1 < K && gc < N) ? B[(size_t)(gk + 1) * N + gc] : 0.f;
            int ln = (col & 15) + (((kk >> 3) & 1) ? 16 : 0);
            int w  = ((kk & 7) >> 1) + ((kk >= 16) ? 4 : 0);
            sBf[col >> 4][ln][w] = f2bf_pack(lo, hi);
        }
        __syncthreads();

        // ---- compute: 1 B fragment x 4 A fragments -> 4 WMMAs ------------
        v8ui bv = *(const v8ui*)(&sBf[wave][lane][0]);
        v16bf b = __builtin_bit_cast(v16bf, bv);

        v8ui a0 = *(const v8ui*)(&sAf[0][lane][0]);
        v8ui a1 = *(const v8ui*)(&sAf[1][lane][0]);
        v8ui a2 = *(const v8ui*)(&sAf[2][lane][0]);
        v8ui a3 = *(const v8ui*)(&sAf[3][lane][0]);

        acc0 = __builtin_amdgcn_wmma_f32_16x16x32_bf16(
                   false, __builtin_bit_cast(v16bf, a0), false, b,
                   (short)0, acc0, false, false);
        acc1 = __builtin_amdgcn_wmma_f32_16x16x32_bf16(
                   false, __builtin_bit_cast(v16bf, a1), false, b,
                   (short)0, acc1, false, false);
        acc2 = __builtin_amdgcn_wmma_f32_16x16x32_bf16(
                   false, __builtin_bit_cast(v16bf, a2), false, b,
                   (short)0, acc2, false, false);
        acc3 = __builtin_amdgcn_wmma_f32_16x16x32_bf16(
                   false, __builtin_bit_cast(v16bf, a3), false, b,
                   (short)0, acc3, false, false);
        __syncthreads();
    }

    // ---- epilogue: VGPR i -> row = i + (lane<16?0:8), col = lane&15 ------
    const int gcol = col0 + wave * 16 + (lane & 15);
    float bv = 0.f;
    if ((flags & 2) && gcol < N) bv = bias[gcol];

    v8f accs[4] = {acc0, acc1, acc2, acc3};
#pragma unroll
    for (int g = 0; g < 4; ++g) {
        int rbase = row0 + g * 16 + ((lane < 16) ? 0 : 8);
#pragma unroll
        for (int i = 0; i < 8; ++i) {
            int r = rbase + i;
            if (r < M && gcol < N) {
                float v = accs[g][i] + bv;
                if (flags & 1) v = fmaxf(v, 0.f);
                C[(size_t)r * N + gcol] = v;
            }
        }
    }
}

// ---------------------------------------------------------------------------
// BatchNorm (training-mode batch stats) — stats per column
// ---------------------------------------------------------------------------
__global__ __launch_bounds__(256)
void bn_stats_kernel(const float* __restrict__ X, int M, int N,
                     float* __restrict__ mean, float* __restrict__ istd)
{
    __shared__ float ss[256], ss2[256];
    int c = blockIdx.x;
    float s = 0.f, s2 = 0.f;
    for (int r = threadIdx.x; r < M; r += 256) {
        float v = X[(size_t)r * N + c];
        s += v; s2 += v * v;
    }
    ss[threadIdx.x] = s; ss2[threadIdx.x] = s2;
    __syncthreads();
    for (int off = 128; off > 0; off >>= 1) {
        if (threadIdx.x < off) {
            ss[threadIdx.x]  += ss[threadIdx.x + off];
            ss2[threadIdx.x] += ss2[threadIdx.x + off];
        }
        __syncthreads();
    }
    if (threadIdx.x == 0) {
        float m   = ss[0] / (float)M;
        float var = ss2[0] / (float)M - m * m;
        mean[c] = m;
        istd[c] = rsqrtf(fmaxf(var, 0.f) + 1e-5f);
    }
}

__global__ void bn_apply_relu_kernel(float* __restrict__ X, int M, int N,
                                     const float* __restrict__ mean,
                                     const float* __restrict__ istd,
                                     const float* __restrict__ gamma,
                                     const float* __restrict__ beta)
{
    int idx = blockIdx.x * blockDim.x + threadIdx.x;
    if (idx >= M * N) return;
    int c = idx % N;
    float v = (X[idx] - mean[c]) * istd[c] * gamma[c] + beta[c];
    X[idx] = fmaxf(v, 0.f);
}

// ---------------------------------------------------------------------------
// Brute-force kNN with LDS candidate tiling, register top-k (bubble insert).
// col0==1 : write self index in column 0 (knn_graph + self loop)
// ---------------------------------------------------------------------------
template<int KQ>
__global__ __launch_bounds__(128)
void knn_kernel(const float* __restrict__ Q, int nq,
                const float* __restrict__ P, int np,
                int* __restrict__ out, int ldo, int col0, int exclude_diag)
{
    __shared__ float sp[128 * 3];
    int q = blockIdx.x * blockDim.x + threadIdx.x;
    bool active = (q < nq);
    float qx = 0.f, qy = 0.f, qz = 0.f;
    if (active) { qx = Q[q*3+0]; qy = Q[q*3+1]; qz = Q[q*3+2]; }

    float bd[KQ]; int bi[KQ];
#pragma unroll
    for (int i = 0; i < KQ; ++i) { bd[i] = 3.4e38f; bi[i] = 0; }

    for (int t0 = 0; t0 < np; t0 += 128) {
        int nt = min(128, np - t0);
        __syncthreads();
        for (int e = threadIdx.x; e < nt * 3; e += 128) sp[e] = P[t0 * 3 + e];
        if (t0 + 128 < np)
            __builtin_prefetch(&P[(t0 + 128) * 3 + threadIdx.x], 0, 0);
        __syncthreads();
        if (active) {
            for (int j = 0; j < nt; ++j) {
                int pj = t0 + j;
                if (exclude_diag && pj == q) continue;
                float dx = qx - sp[j*3+0];
                float dy = qy - sp[j*3+1];
                float dz = qz - sp[j*3+2];
                float d = dx*dx + dy*dy + dz*dz;
                if (d < bd[KQ - 1]) {
                    bd[KQ - 1] = d; bi[KQ - 1] = pj;
#pragma unroll
                    for (int ii = KQ - 1; ii > 0; --ii) {
                        if (bd[ii] < bd[ii - 1]) {
                            float tf = bd[ii]; bd[ii] = bd[ii-1]; bd[ii-1] = tf;
                            int   ti = bi[ii]; bi[ii] = bi[ii-1]; bi[ii-1] = ti;
                        }
                    }
                }
            }
        }
    }
    if (active) {
        if (col0) out[(size_t)q * ldo] = q;
#pragma unroll
        for (int i = 0; i < KQ; ++i) out[(size_t)q * ldo + col0 + i] = bi[i];
    }
}

// ---------------------------------------------------------------------------
// Farthest-point sampling (single block, LDS mind array + argmax reduction)
// ---------------------------------------------------------------------------
__global__ __launch_bounds__(1024)
void fps_kernel(const float* __restrict__ pos, int N, int nsamp,
                int* __restrict__ out)
{
    __shared__ float mind[8192];
    __shared__ float rv[32];
    __shared__ int   ri[32];
    __shared__ int   s_last;
    int tid = threadIdx.x;
    for (int i = tid; i < N; i += 1024) mind[i] = 3.4e38f;
    int last = 0;
    __syncthreads();

    for (int s = 0; s < nsamp; ++s) {
        if (tid == 0) out[s] = last;
        float lx = pos[last*3+0], ly = pos[last*3+1], lz = pos[last*3+2];
        float best = -1.f; int bidx = 0;
        for (int i = tid; i < N; i += 1024) {
            float dx = pos[i*3+0] - lx;
            float dy = pos[i*3+1] - ly;
            float dz = pos[i*3+2] - lz;
            float d = dx*dx + dy*dy + dz*dz;
            float m = fminf(mind[i], d);
            mind[i] = m;
            if (m > best) { best = m; bidx = i; }
        }
        for (int off = 16; off > 0; off >>= 1) {
            float ov = __shfl_down(best, off, 32);
            int   oi = __shfl_down(bidx, off, 32);
            if (ov > best) { best = ov; bidx = oi; }
        }
        int wid = tid >> 5;
        if ((tid & 31) == 0) { rv[wid] = best; ri[wid] = bidx; }
        __syncthreads();
        if (tid == 0) {
            float bb = rv[0]; int bj = ri[0];
            for (int w = 1; w < 32; ++w)
                if (rv[w] > bb) { bb = rv[w]; bj = ri[w]; }
            s_last = bj;
        }
        __syncthreads();
        last = s_last;
    }
}

// ---------------------------------------------------------------------------
// Misc gather / edge kernels
// ---------------------------------------------------------------------------
__global__ void gather_pos_kernel(const float* __restrict__ pos,
                                  const int* __restrict__ idx, int n,
                                  float* __restrict__ out)
{
    int i = blockIdx.x * blockDim.x + threadIdx.x;
    if (i >= n) return;
    int s = idx[i];
    out[i*3+0] = pos[s*3+0];
    out[i*3+1] = pos[s*3+1];
    out[i*3+2] = pos[s*3+2];
}

__global__ void edge_posdiff_kernel(const float* __restrict__ pos,
                                    const int* __restrict__ nbr,
                                    int N, int KK, float* __restrict__ out)
{
    int e = blockIdx.x * blockDim.x + threadIdx.x;
    if (e >= N * KK) return;
    int i = e / KK;
    int j = nbr[e];
    out[e*3+0] = pos[i*3+0] - pos[j*3+0];
    out[e*3+1] = pos[i*3+1] - pos[j*3+1];
    out[e*3+2] = pos[i*3+2] - pos[j*3+2];
}

__global__ void edge_combine_kernel(const float* __restrict__ q,
                                    const float* __restrict__ s,
                                    const float* __restrict__ delta,
                                    const int* __restrict__ nbr,
                                    int N, int KK, int C,
                                    float* __restrict__ out)
{
    int idx = blockIdx.x * blockDim.x + threadIdx.x;
    if (idx >= N * KK * C) return;
    int e = idx / C;
    int c = idx - e * C;
    int i = e / KK;
    int j = nbr[e];
    out[idx] = q[(size_t)i * C + c] - s[(size_t)j * C + c] + delta[idx];
}

__global__ void gather_max_kernel(const float* __restrict__ h,
                                  const int* __restrict__ nbr,
                                  int nsub, int k, int C,
                                  float* __restrict__ out)
{
    int idx = blockIdx.x * blockDim.x + threadIdx.x;
    if (idx >= nsub * C) return;
    int i = idx / C;
    int c = idx - i * C;
    float m = -3.4e38f;
    for (int j = 0; j < k; ++j)
        m = fmaxf(m, h[(size_t)nbr[i * k + j] * C + c]);
    out[idx] = m;
}

// Per-point, per-channel softmax over (k+1) neighbors, then aggregate
__global__ __launch_bounds__(128)
void attn_agg_kernel(const float* __restrict__ alpha,
                     const float* __restrict__ v,
                     const float* __restrict__ delta,
                     const int* __restrict__ nbr,
                     int N, int KK, int C, float* __restrict__ out)
{
    int i = blockIdx.x;
    for (int c = threadIdx.x; c < C; c += blockDim.x) {
        float m = -3.4e38f;
        for (int j = 0; j < KK; ++j)
            m = fmaxf(m, alpha[(size_t)(i * KK + j) * C + c]);
        float ssum = 0.f, acc = 0.f;
        for (int j = 0; j < KK; ++j) {
            size_t e = (size_t)(i * KK + j);
            float a = __expf(alpha[e * C + c] - m);
            ssum += a;
            acc += a * (v[(size_t)nbr[e] * C + c] + delta[e * C + c]);
        }
        out[(size_t)i * C + c] = acc / ssum;
    }
}

// out[i,c] = base[i,c] + sum_k w_k * xs[idx[i,k],c] / sum_k w_k
__global__ __launch_bounds__(128)
void interp_add_kernel(const float* __restrict__ base,
                       const float* __restrict__ xs,
                       const int* __restrict__ idx,
                       const float* __restrict__ pos,
                       const float* __restrict__ pos_sub,
                       int N, int C, float* __restrict__ out)
{
    int i = blockIdx.x;
    int i0 = idx[i*3+0], i1 = idx[i*3+1], i2 = idx[i*3+2];
    float px = pos[i*3+0], py = pos[i*3+1], pz = pos[i*3+2];
    float w[3]; int ii[3] = {i0, i1, i2};
#pragma unroll
    for (int k = 0; k < 3; ++k) {
        float dx = px - pos_sub[ii[k]*3+0];
        float dy = py - pos_sub[ii[k]*3+1];
        float dz = pz - pos_sub[ii[k]*3+2];
        float d2 = dx*dx + dy*dy + dz*dz;
        w[k] = 1.f / fmaxf(d2, 1e-16f);
    }
    float wsum = w[0] + w[1] + w[2];
    for (int c = threadIdx.x; c < C; c += blockDim.x) {
        float v = w[0] * xs[(size_t)i0 * C + c]
                + w[1] * xs[(size_t)i1 * C + c]
                + w[2] * xs[(size_t)i2 * C + c];
        out[(size_t)i * C + c] = base[(size_t)i * C + c] + v / wsum;
    }
}

__global__ void row_softmax_kernel(const float* __restrict__ in, int M, int N,
                                   float* __restrict__ out)
{
    int r = blockIdx.x * blockDim.x + threadIdx.x;
    if (r >= M) return;
    float m = -3.4e38f;
    for (int c = 0; c < N; ++c) m = fmaxf(m, in[(size_t)r * N + c]);
    float s = 0.f;
    for (int c = 0; c < N; ++c) s += __expf(in[(size_t)r * N + c] - m);
    float inv = 1.f / s;
    for (int c = 0; c < N; ++c)
        out[(size_t)r * N + c] = __expf(in[(size_t)r * N + c] - m) * inv;
}

// ---------------------------------------------------------------------------
// Host-side orchestration
// ---------------------------------------------------------------------------
namespace {

struct Lin { const float* W; const float* b; };
struct BN  { const float* beta; const float* gamma; Lin lin; };
struct TB  {
    Lin attn0, attn1;
    const float* Wlin;
    const float* Wdst;
    const float* Wsrc;
    Lin pos0, pos1;
    Lin lin_in, lin_out;
};

struct Arena {
    char* base; size_t off, cap;
    void init(void* p, size_t c) { base = (char*)p; off = 0; cap = c; }
    void* alloc(size_t bytes) {
        off = (off + 255) & ~(size_t)255;
        void* p = base + off;
        off += bytes;
        if (off > cap) off = cap;
        return p;
    }
    float* allocF(size_t n) { return (float*)alloc(n * sizeof(float)); }
    int*   allocI(size_t n) { return (int*)alloc(n * sizeof(int)); }
    size_t mark() const { return off; }
    void   reset(size_t m) { off = m; }
};

inline int cdiv(int a, int b) { return (a + b - 1) / b; }

void gemm(hipStream_t st, const float* A, const float* W, const float* bias,
          float* C, int M, int K, int N, bool relu)
{
    dim3 g(cdiv(N, 64), cdiv(M, 64));
    int flags = (relu ? 1 : 0) | (bias ? 2 : 0);
    gemm_wmma_kernel<<<g, 128, 0, st>>>(A, W, bias, C, M, K, N, flags);
}

void bn_layer(hipStream_t st, Arena& S, const BN& l, const float* in,
              float* out, int M, int Kin, int Kout)
{
    size_t m = S.mark();
    gemm(st, in, l.lin.W, l.lin.b, out, M, Kin, Kout, false);
    float* mean = S.allocF(Kout);
    float* istd = S.allocF(Kout);
    bn_stats_kernel<<<Kout, 256, 0, st>>>(out, M, Kout, mean, istd);
    int tot = M * Kout;
    bn_apply_relu_kernel<<<cdiv(tot, 256), 256, 0, st>>>(
        out, M, Kout, mean, istd, l.gamma, l.beta);
    S.reset(m);
}

float* run_tblock(hipStream_t st, Arena& P, Arena& S, const TB& tb,
                  const float* x, const float* pos, const int* nbr,
                  int N, int Cin, int Cout)
{
    const int KK = 17;
    const int E  = N * KK;
    size_t m0 = S.mark();

    float* x1 = S.allocF((size_t)N * Cin);
    gemm(st, x, tb.lin_in.W, tb.lin_in.b, x1, N, Cin, Cin, true);

    float* q = S.allocF((size_t)N * Cout);
    float* s = S.allocF((size_t)N * Cout);
    float* v = S.allocF((size_t)N * Cout);
    gemm(st, x1, tb.Wdst, nullptr, q, N, Cin, Cout, false);
    gemm(st, x1, tb.Wsrc, nullptr, s, N, Cin, Cout, false);
    gemm(st, x1, tb.Wlin, nullptr, v, N, Cin, Cout, false);

    float* delta = S.allocF((size_t)E * Cout);
    float* agg   = S.allocF((size_t)N * Cout);

    {   // pos_nn: (pos_i - pos_j) -> 64 -> Cout (relu after each layer)
        size_t m1 = S.mark();
        float* pd = S.allocF((size_t)E * 3);
        float* h  = S.allocF((size_t)E * 64);
        edge_posdiff_kernel<<<cdiv(E, 256), 256, 0, st>>>(pos, nbr, N, KK, pd);
        gemm(st, pd, tb.pos0.W, tb.pos0.b, h, E, 3, 64, true);
        gemm(st, h, tb.pos1.W, tb.pos1.b, delta, E, 64, Cout, true);
        S.reset(m1);
    }
    {   // attn_nn: (q_i - s_j + delta) -> 64 -> Cout, softmax, aggregate
        size_t m1 = S.mark();
        float* t  = S.allocF((size_t)E * Cout);
        float* h2 = S.allocF((size_t)E * 64);
        int tot = E * Cout;
        edge_combine_kernel<<<cdiv(tot, 256), 256, 0, st>>>(
            q, s, delta, nbr, N, KK, Cout, t);
        gemm(st, t, tb.attn0.W, tb.attn0.b, h2, E, Cout, 64, true);
        gemm(st, h2, tb.attn1.W, tb.attn1.b, t, E, 64, Cout, true);
        attn_agg_kernel<<<N, 128, 0, st>>>(t, v, delta, nbr, N, KK, Cout, agg);
        S.reset(m1);
    }

    float* x2 = P.allocF((size_t)N * Cout);
    gemm(st, agg, tb.lin_out.W, tb.lin_out.b, x2, N, Cout, Cout, true);
    S.reset(m0);
    return x2;
}

} // namespace

// ---------------------------------------------------------------------------
extern "C" void kernel_launch(void* const* d_in, const int* in_sizes, int n_in,
                              void* d_out, int out_size, void* d_ws, size_t ws_size,
                              hipStream_t stream)
{
    (void)in_sizes; (void)out_size;
    hipStream_t st = stream;

    // ---- parse pytree leaves (sorted dict key order) ----
    int cur = 0;
    auto F = [&]() -> const float* { return (const float*)d_in[cur++]; };
    auto rdLin = [&]() { Lin l; l.W = F(); l.b = F(); return l; };
    auto rdBN  = [&]() { BN b; b.beta = F(); b.gamma = F();
                         b.lin.W = F(); b.lin.b = F(); return b; };
    auto rdTB  = [&]() {
        TB t;
        t.attn0 = rdLin(); t.attn1 = rdLin();
        t.Wlin = F(); t.Wdst = F(); t.Wsrc = F();
        t.pos0 = rdLin(); t.pos1 = rdLin();
        t.lin_in = rdLin(); t.lin_out = rdLin();
        return t;
    };

    BN  down_td[4]; TB down_tf[4];
    for (int i = 0; i < 4; ++i) { down_td[i] = rdBN(); down_tf[i] = rdTB(); }
    BN  mlp_input  = rdBN();
    Lin mlp_summit = rdLin();
    Lin out1 = rdLin(), out2 = rdLin(), out3 = rdLin();
    TB  tf_input  = rdTB();
    TB  tf_summit = rdTB();
    BN  up_mlp[4], up_sub[4]; TB up_tf[4];
    for (int i = 0; i < 4; ++i) { up_mlp[i] = rdBN(); up_sub[i] = rdBN(); up_tf[i] = rdTB(); }

    const float* pos0 = (const float*)d_in[n_in - 2];   // "pos"
    const float* x_in = (const float*)d_in[n_in - 1];   // "x"

    Arena P, S;
    size_t pcap = (ws_size / 4) & ~(size_t)255;
    P.init(d_ws, pcap);
    S.init((char*)d_ws + pcap, ws_size - pcap);

    const int dims[5] = {32, 64, 128, 256, 512};
    const int Ns[5]   = {8192, 2048, 512, 128, 32};

    struct Skip { float* x; const float* pos; const int* nbr; };
    Skip skips[5];

    // ---- input MLP + first transformer block ----
    int* nbr0 = P.allocI((size_t)Ns[0] * 17);
    knn_kernel<16><<<cdiv(Ns[0], 128), 128, 0, st>>>(
        pos0, Ns[0], pos0, Ns[0], nbr0, 17, 1, 1);

    float* x0 = P.allocF((size_t)Ns[0] * dims[0]);
    bn_layer(st, S, mlp_input, x_in, x0, Ns[0], 3, dims[0]);

    float* x = run_tblock(st, P, S, tf_input, x0, pos0, nbr0, Ns[0], dims[0], dims[0]);
    skips[0] = {x, pos0, nbr0};
    const float* cur_pos = pos0;

    // ---- encoder ----
    for (int lev = 0; lev < 4; ++lev) {
        int N = Ns[lev], Nsub = Ns[lev + 1];
        int Cn = dims[lev + 1];

        int* idc = P.allocI(Nsub);
        fps_kernel<<<1, 1024, 0, st>>>(cur_pos, N, Nsub, idc);

        float* sub_pos = P.allocF((size_t)Nsub * 3);
        gather_pos_kernel<<<cdiv(Nsub, 128), 128, 0, st>>>(cur_pos, idc, Nsub, sub_pos);

        size_t m = S.mark();
        int* nbrd = S.allocI((size_t)Nsub * 16);
        knn_kernel<16><<<cdiv(Nsub, 128), 128, 0, st>>>(
            sub_pos, Nsub, cur_pos, N, nbrd, 16, 0, 0);

        float* h = S.allocF((size_t)N * Cn);
        bn_layer(st, S, down_td[lev], x, h, N, dims[lev], Cn);

        float* xsub = P.allocF((size_t)Nsub * Cn);
        gather_max_kernel<<<cdiv(Nsub * Cn, 256), 256, 0, st>>>(
            h, nbrd, Nsub, 16, Cn, xsub);
        S.reset(m);

        int* nbrg = P.allocI((size_t)Nsub * 17);
        knn_kernel<16><<<cdiv(Nsub, 128), 128, 0, st>>>(
            sub_pos, Nsub, sub_pos, Nsub, nbrg, 17, 1, 1);

        x = run_tblock(st, P, S, down_tf[lev], xsub, sub_pos, nbrg, Nsub, Cn, Cn);
        skips[lev + 1] = {x, sub_pos, nbrg};
        cur_pos = sub_pos;
    }

    // ---- summit ----
    {
        float* xm = P.allocF((size_t)Ns[4] * dims[4]);
        gemm(st, x, mlp_summit.W, mlp_summit.b, xm, Ns[4], dims[4], dims[4], true);
        x = run_tblock(st, P, S, tf_summit, xm, skips[4].pos, skips[4].nbr,
                       Ns[4], dims[4], dims[4]);
    }

    // ---- decoder ----
    for (int i = 0; i < 4; ++i) {
        int li = 3 - i;
        int N_cur = Ns[li + 1], C_cur = dims[li + 1];
        int N_skip = Ns[li],    C_skip = dims[li];
        const Skip& sk = skips[li];
        const float* pos_sub = skips[li + 1].pos;

        size_t m = S.mark();
        float* xs = S.allocF((size_t)N_cur * C_skip);
        bn_layer(st, S, up_sub[li], x, xs, N_cur, C_cur, C_skip);

        float* base = S.allocF((size_t)N_skip * C_skip);
        bn_layer(st, S, up_mlp[li], sk.x, base, N_skip, C_skip, C_skip);

        int* idx3 = S.allocI((size_t)N_skip * 3);
        knn_kernel<3><<<cdiv(N_skip, 128), 128, 0, st>>>(
            sk.pos, N_skip, pos_sub, N_cur, idx3, 3, 0, 0);

        float* xnew = P.allocF((size_t)N_skip * C_skip);
        interp_add_kernel<<<N_skip, 128, 0, st>>>(
            base, xs, idx3, sk.pos, pos_sub, N_skip, C_skip, xnew);
        S.reset(m);

        x = run_tblock(st, P, S, up_tf[li], xnew, sk.pos, sk.nbr,
                       N_skip, C_skip, C_skip);
    }

    // ---- head ----
    {
        size_t m = S.mark();
        float* h1 = S.allocF((size_t)Ns[0] * 64);
        float* h2 = S.allocF((size_t)Ns[0] * 64);
        float* o  = S.allocF((size_t)Ns[0] * 13);
        gemm(st, x,  out1.W, out1.b, h1, Ns[0], dims[0], 64, true);
        gemm(st, h1, out2.W, out2.b, h2, Ns[0], 64, 64, true);
        gemm(st, h2, out3.W, out3.b, o,  Ns[0], 64, 13, false);
        row_softmax_kernel<<<cdiv(Ns[0], 128), 128, 0, st>>>(
            o, Ns[0], 13, (float*)d_out);
        S.reset(m);
    }
}
